// Criterion_38044820308458
// MI455X (gfx1250) — compile-verified, matrix operand order checked by reading, not run on hardware
//
#include <hip/hip_runtime.h>

typedef __bf16 bf16;
typedef __attribute__((ext_vector_type(16))) __bf16 v16bf;
typedef __attribute__((ext_vector_type(8)))  __bf16 v8bf;
typedef __attribute__((ext_vector_type(8)))  float  v8f;

#define DIM      768
#define NROWS    8192
#define NTILES   512              // 8192 / 16 column tiles
#define KF       24               // 768 / 32 WMMA K-steps
#define TEMP_INV 20.0f            // 1 / 0.05
#define EPSN     1e-8f
#define WPB      4                // waves per block
#define ROWS_PB  (WPB * 16)       // 64 rows per block
#define NBLOCKS  (NROWS / ROWS_PB)
#define TILE_BYTES 24576          // 16 rows * 768 * 2 B per B tile (global)
#define CHUNKS_PER_THREAD 12      // per-thread 16B chunks of a tile copy
#define LDS_ROW    776            // 768 + 8 bf16 pad -> stride 388 dw == 4 mod 64
#define LDS_TILE_B (16 * LDS_ROW * 2)          // 24832 B per LDS tile
#define SMEM_MAIN  (3 * LDS_TILE_B)            // triple buffer: 74496 B
#define SMEM_BYTES (SMEM_MAIN + WPB * 16 * 4)  // + s_tgt

// ---------------------------------------------------------------------------
// One 16x32 bf16 WMMA fragment for this lane from a row-major row pointer.
// ISA 16-bit fragment layout (wave32):
//   lanes 0-15  : K = k0+0..7  and K = k0+16..23
//   lanes 16-31 : K = k0+8..15 and K = k0+24..31
// ---------------------------------------------------------------------------
__device__ __forceinline__ v16bf load_frag(const bf16* __restrict__ rowp,
                                           int k0, int half) {
  union { v16bf v; v8bf h[2]; } f;
  f.h[0] = *(const v8bf*)(rowp + k0 + half * 8);
  f.h[1] = *(const v8bf*)(rowp + k0 + 16 + half * 8);
  return f.v;
}

// ---------------------------------------------------------------------------
// Kernel 1: row L2-normalize with eps clamp, emit bf16 xn (12 MB, L2-resident)
// ---------------------------------------------------------------------------
__global__ __launch_bounds__(256) void normalize_kernel(
    const float* __restrict__ x, bf16* __restrict__ xn) {
  const int row = blockIdx.x;
  const int tid = threadIdx.x;
  const float* xr = x + (size_t)row * DIM;
  float v0 = xr[tid], v1 = xr[tid + 256], v2 = xr[tid + 512];
  float ss = v0 * v0 + v1 * v1 + v2 * v2;
#pragma unroll
  for (int s = 1; s < 32; s <<= 1) ss += __shfl_xor(ss, s, 32);
  __shared__ float wsum[8];
  if ((tid & 31) == 0) wsum[tid >> 5] = ss;
  __syncthreads();
  float tot = 0.f;
#pragma unroll
  for (int w = 0; w < 8; ++w) tot += wsum[w];
  const float inv = 1.0f / fmaxf(sqrtf(tot), EPSN);
  bf16* outr = xn + (size_t)row * DIM;
  outr[tid]       = (bf16)(v0 * inv);
  outr[tid + 256] = (bf16)(v1 * inv);
  outr[tid + 512] = (bf16)(v2 * inv);
}

// ---------------------------------------------------------------------------
// Kernel 2: fused sim-GEMM + softmax + NLL.
// Block = 64 rows (4 waves x 16 rows). B tiles DMA'd global->LDS with async
// loads (ASYNCcnt), TRIPLE-buffered (one barrier/tile, 2 tiles of DMA lead).
// LDS rows padded to 776 bf16 so ds_load_b128 fragment reads are bank-optimal.
// A strip stays in 192 VGPRs per wave; 24 WMMAs per tile.
// ---------------------------------------------------------------------------
__global__ __launch_bounds__(128) void loss_kernel(
    const bf16* __restrict__ xn, float* __restrict__ out) {
  extern __shared__ char smem[];
  bf16*  s_b   = (bf16*)smem;                 // [3][16][LDS_ROW]
  float* s_tgt = (float*)(smem + SMEM_MAIN);  // [WPB][16]

  const int tid    = threadIdx.x;
  const int wave   = tid >> 5;
  const int lane   = tid & 31;
  const int lane16 = lane & 15;
  const int half   = lane >> 4;

  if (lane < 16) s_tgt[wave * 16 + lane] = 0.f;

  const int rowbase   = blockIdx.x * ROWS_PB + wave * 16;
  const int diag_tile = rowbase >> 4;

  // A fragments in registers: row = rowbase + lane16 (M = lane%16).
  const bf16* arow = xn + (size_t)(rowbase + lane16) * DIM;
  v16bf afrag[KF];
#pragma unroll
  for (int kf = 0; kf < KF; ++kf) afrag[kf] = load_frag(arow, kf * 32, half);

  // Async copy: 8 threads per tile row, 12 chunks of 16 B at stride 128 B.
  // Same immediate offset advances both LDS (padded row) and global (dense
  // row) sides without crossing a row boundary. LDS addr = low 32 ptr bits.
  const int crow = tid >> 3, cpos = (tid & 7) * 16;
  const unsigned long long gcopy =
      (unsigned long long)(uintptr_t)xn + (unsigned)(crow * 1536 + cpos);
  const unsigned lcopy =
      (unsigned)(uintptr_t)smem + (unsigned)(crow * (LDS_ROW * 2) + cpos);

  auto issue_copy = [&](int tile, int buf) {
    unsigned long long g = gcopy + (unsigned long long)tile * TILE_BYTES;
    unsigned           l = lcopy + (unsigned)buf * LDS_TILE_B;
#pragma unroll
    for (int c = 0; c < CHUNKS_PER_THREAD; ++c)
      asm volatile("global_load_async_to_lds_b128 %0, %1, off offset:%2"
                   :: "v"(l), "v"(g), "i"(c * 128) : "memory");
  };

  float esum[8];
#pragma unroll
  for (int r = 0; r < 8; ++r) esum[r] = 0.f;

  issue_copy(0, 0);
  issue_copy(1, 1);

  for (int t = 0; t < NTILES; ++t) {
    // Drain own shares of tile t (async loads complete in order).
    if (t + 1 < NTILES) asm volatile("s_wait_asynccnt 12" ::: "memory");
    else                asm volatile("s_wait_asynccnt 0"  ::: "memory");
    __syncthreads();  // all waves' shares visible; old buffer reads finished
    if (t + 2 < NTILES) issue_copy(t + 2, (t + 2) % 3);

    const bf16* bbase = s_b + (t % 3) * (16 * LDS_ROW) + lane16 * LDS_ROW;

    // Depth-1 ds_load pipeline: fragment kf+1 loads issue before wmma(kf).
    v8f acc = {};
    v16bf bcur = load_frag(bbase, 0, half);
#pragma unroll
    for (int kf = 0; kf < KF; ++kf) {
      v16bf bnext = bcur;
      if (kf + 1 < KF) bnext = load_frag(bbase, (kf + 1) * 32, half);
      acc = __builtin_amdgcn_wmma_f32_16x16x32_bf16(
          /*neg_a=*/false, afrag[kf], /*neg_b=*/false, bcur,
          /*c_mod=*/(short)0, acc, /*reuse_a=*/false, /*reuse_b=*/false);
      bcur = bnext;
    }

    // C layout: acc[r] = sim(M = r + 8*half, N = lane16) for this tile.
    const int colbase = t << 4;
    if (t != diag_tile) {
#pragma unroll
      for (int r = 0; r < 8; ++r) esum[r] += __expf(acc[r] * TEMP_INV);
    } else {
      const int gcol = colbase + lane16;
#pragma unroll
      for (int r = 0; r < 8; ++r) {
        const int m    = r + 8 * half;
        const int grow = rowbase + m;
        const float logit = acc[r] * TEMP_INV;
        if (gcol == (grow ^ 1)) s_tgt[wave * 16 + m] = logit;
        if (gcol != grow) esum[r] += __expf(logit);  // diag term is exactly 0
      }
    }
  }

  // Full row exp-sums: reduce across the 16 lanes sharing the same 8 rows.
#pragma unroll
  for (int s = 1; s < 16; s <<= 1) {
#pragma unroll
    for (int r = 0; r < 8; ++r) esum[r] += __shfl_xor(esum[r], s, 16);
  }
  __syncthreads();

  if (lane16 == 0) {  // lane 0 -> rows 0..7, lane 16 -> rows 8..15
    float part = 0.f;
#pragma unroll
    for (int r = 0; r < 8; ++r)
      part += __logf(esum[r]) - s_tgt[wave * 16 + r + 8 * half];
    atomicAdd(out, part * (1.0f / (float)NROWS));
  }
}

// ---------------------------------------------------------------------------
extern "C" void kernel_launch(void* const* d_in, const int* in_sizes, int n_in,
                              void* d_out, int out_size, void* d_ws, size_t ws_size,
                              hipStream_t stream) {
  (void)in_sizes; (void)n_in; (void)out_size; (void)ws_size;
  const float* x   = (const float*)d_in[0];
  bf16*        xn  = (bf16*)d_ws;   // 8192*768*2 = 12 MB scratch
  float*       out = (float*)d_out;

  hipMemsetAsync(d_out, 0, sizeof(float), stream);  // capture-safe zero
  normalize_kernel<<<NROWS, 256, 0, stream>>>(x, xn);
  loss_kernel<<<NBLOCKS, 128, SMEM_BYTES, stream>>>(xn, out);
}
